// StructuralAttentionLayer_30511447671553
// MI455X (gfx1250) — matrix operand order repaired and checked by reading, not compiled, not run on hardware
//
#include <hip/hip_runtime.h>
#include <math.h>

// Problem constants (match reference)
#define NN 4096   // nodes
#define FF 512    // input dim
#define DD 512    // output dim (H*dh)
#define HH 8      // heads
#define DH 64     // per-head dim

typedef __attribute__((ext_vector_type(16))) __bf16        v16bf;
typedef __attribute__((ext_vector_type(8)))  float         v8f;
typedef __attribute__((ext_vector_type(4)))  unsigned int  v4u;
typedef __attribute__((ext_vector_type(8)))  unsigned int  v8u;
typedef __attribute__((ext_vector_type(4)))  float         v4fl;

__device__ __forceinline__ unsigned int f32_to_bf16_rne(float f) {
  unsigned int u = __float_as_uint(f);
  u += 0x7fffu + ((u >> 16) & 1u);
  return u >> 16;
}

// Load a 16x32 bf16 WMMA operand fragment: 16 contiguous bf16 from p0 (elements
// 0..7) and p1 (elements 8..15), per the CDNA5 16-bit A/B VGPR layouts.
__device__ __forceinline__ v16bf load_frag16(const unsigned short* p0,
                                             const unsigned short* p1) {
  v4u lo = *reinterpret_cast<const v4u*>(p0);
  v4u hi = *reinterpret_cast<const v4u*>(p1);
  v8u w = {lo[0], lo[1], lo[2], lo[3], hi[0], hi[1], hi[2], hi[3]};
  return __builtin_bit_cast(v16bf, w);
}

// ---------------------------------------------------------------------------
// Kernel 1a: pack x (f32 row-major [N,F]) -> bf16 row-major
__global__ void __launch_bounds__(256) k_pack_x(const float* __restrict__ x,
                                                unsigned short* __restrict__ xb) {
  int i = blockIdx.x * 256 + threadIdx.x;
  if (i < NN * FF) xb[i] = (unsigned short)f32_to_bf16_rne(x[i]);
}

// Kernel 1b: pack W [H,F,dh] f32 -> Wt [H,dh,F] bf16 (transposed so B-operand
// fragments along K are contiguous)
__global__ void __launch_bounds__(256) k_pack_w(const float* __restrict__ W,
                                                unsigned short* __restrict__ wbt) {
  int i = blockIdx.x * 256 + threadIdx.x;
  if (i >= HH * DH * FF) return;
  int f = i % FF;
  int r = i / FF;
  int dd = r % DH;
  int h  = r / DH;
  wbt[i] = (unsigned short)f32_to_bf16_rne(W[((size_t)h * FF + f) * DH + dd]);
}

// ---------------------------------------------------------------------------
// Kernel 2: seq_fts = x . W[h]  via v_wmma_f32_16x16x32_bf16.
// Grid: N/16 row-blocks; block: 256 threads = 8 waves, wave w == head w
// (x-tile rows shared across all 8 waves through L0).
// Each wave computes a 16x64 f32 tile (4 C tiles), K = 512 in 16 steps.
// Writes seq_fts f32 [H,N,64] and its bf16 transpose [H,64,N] (V operand of
// the aggregation matmul).
__global__ void __launch_bounds__(256) k_seqfts(const unsigned short* __restrict__ xb,
                                                const unsigned short* __restrict__ wbt,
                                                float* __restrict__ seq,
                                                unsigned short* __restrict__ seqbt) {
  const int base = blockIdx.x * 16;
  const int h    = threadIdx.x >> 5;
  const int lane = threadIdx.x & 31;
  const bool hi  = lane >= 16;
  const int c    = lane & 15;
  const int aoff = hi ? 8 : 0;    // A-frag K sub-offset (16-bit layout)
  const int boff = hi ? 8 : 0;    // B-frag K sub-offset

  v8f acc[4] = {};
  const unsigned short* xrow = xb + (size_t)(base + c) * FF;

  for (int k = 0; k < FF; k += 32) {
    v16bf a = load_frag16(xrow + k + aoff, xrow + k + 16 + aoff);
#pragma unroll
    for (int t = 0; t < 4; ++t) {
      const unsigned short* wp =
          wbt + (size_t)(h * DH + t * 16 + c) * FF + k + boff;
      v16bf b = load_frag16(wp, wp + 16);
      acc[t] = __builtin_amdgcn_wmma_f32_16x16x32_bf16(
          false, a, false, b, (short)0, acc[t], false, false);
    }
  }

  // C layout: VGPR e -> row e (lanes 0-15) / row e+8 (lanes 16-31), col = c.
#pragma unroll
  for (int t = 0; t < 4; ++t) {
#pragma unroll
    for (int e = 0; e < 8; ++e) {
      int n  = base + e + (hi ? 8 : 0);
      int dd = t * 16 + c;
      float v = acc[t][e];
      seq[((size_t)h * NN + n) * DH + dd] = v;
      seqbt[(size_t)(h * DH + dd) * NN + n] = (unsigned short)f32_to_bf16_rne(v);
    }
  }
}

// ---------------------------------------------------------------------------
// Kernel 3: f1/f2 per (h,n): dot of seq_fts row with a1/a2 plus bias.
__global__ void __launch_bounds__(256) k_f1f2(const float* __restrict__ seq,
                                              const float* __restrict__ a1,
                                              const float* __restrict__ b1,
                                              const float* __restrict__ a2,
                                              const float* __restrict__ b2,
                                              float* __restrict__ f1,
                                              float* __restrict__ f2) {
  int i = blockIdx.x * 256 + threadIdx.x;
  if (i >= HH * NN) return;
  int h = i / NN;
  const float* s  = seq + (size_t)i * DH;
  const float* p1 = a1 + h * DH;
  const float* p2 = a2 + h * DH;
  float s1 = 0.f, s2 = 0.f;
#pragma unroll
  for (int q = 0; q < DH / 4; ++q) {
    v4fl v  = *reinterpret_cast<const v4fl*>(s + 4 * q);
    v4fl w1 = *reinterpret_cast<const v4fl*>(p1 + 4 * q);
    v4fl w2 = *reinterpret_cast<const v4fl*>(p2 + 4 * q);
#pragma unroll
    for (int e = 0; e < 4; ++e) { s1 += v[e] * w1[e]; s2 += v[e] * w2[e]; }
  }
  f1[i] = s1 + b1[h];
  f2[i] = s2 + b2[h];
}

// ---------------------------------------------------------------------------
// Kernel 4 (softmax pass A): per (row i, head h) compute masked row max m and
// 1/sum of exp. Grid: N blocks; block 256 = 8 waves, wave w == head w; adj row
// is read by all 8 waves but stays hot in L0/L2.
__global__ void __launch_bounds__(256) k_rowstats(const float* __restrict__ adj,
                                                  const float* __restrict__ f1,
                                                  const float* __restrict__ f2,
                                                  float* __restrict__ m,
                                                  float* __restrict__ linv) {
  const int i    = blockIdx.x;
  const int h    = threadIdx.x >> 5;
  const int lane = threadIdx.x & 31;
  const float* arow = adj + (size_t)i * NN;
  const float* f2h  = f2 + h * NN;
  const float  fi   = f1[h * NN + i];

  float mx = -1e30f;
  for (int j = lane; j < NN; j += 32) {
    float a = arow[j];
    if (a != 0.0f) {
      float s = fi + f2h[j];
      s = s > 0.0f ? s : 0.2f * s;   // leaky relu 0.2
      mx = fmaxf(mx, s);
    }
  }
#pragma unroll
  for (int off = 16; off; off >>= 1) mx = fmaxf(mx, __shfl_xor(mx, off, 32));

  float sum = 0.f;
  for (int j = lane; j < NN; j += 32) {
    float a = arow[j];
    if (a != 0.0f) {
      float s = fi + f2h[j];
      s = s > 0.0f ? s : 0.2f * s;
      sum += __expf(s - mx);
    }
  }
#pragma unroll
  for (int off = 16; off; off >>= 1) sum += __shfl_xor(sum, off, 32);

  if (lane == 0) {
    m[h * NN + i]    = mx;
    linv[h * NN + i] = 1.0f / sum;   // self-loop guarantees sum > 0
  }
}

// ---------------------------------------------------------------------------
// Kernel 5 (softmax pass B + aggregation): out = elu((P/l) . seq_fts).
// Grid: (N/128, H). One workgroup = ONE head over 128 rows; its 8 waves cover
// 8 consecutive 16-row blocks. All 8 waves load the SAME B (seq_fts^T)
// fragments each j-tile, so the 64KB WGP$ absorbs the 8x redundancy and L2
// traffic on seq_fts drops 8x (1GB -> 128MB). A periodic __syncthreads keeps
// the waves within a few tiles of each other so the L0 sharing holds.
// P is built in registers directly in the A-fragment layout (scores are
// rank-1 over the adjacency mask, no QK^T needed).
__global__ void __launch_bounds__(256) k_aggregate(const float* __restrict__ adj,
                                                   const float* __restrict__ f1,
                                                   const float* __restrict__ f2,
                                                   const float* __restrict__ m,
                                                   const float* __restrict__ linv,
                                                   const unsigned short* __restrict__ seqbt,
                                                   float* __restrict__ out) {
  const int wave = threadIdx.x >> 5;
  const int h    = blockIdx.y;
  const int base = blockIdx.x * 128 + wave * 16;
  const int lane = threadIdx.x & 31;
  const bool hi  = lane >= 16;
  const int c    = lane & 15;
  const int row  = base + c;            // A-fragment row for this lane
  const int joff = hi ? 8 : 0;          // K sub-offset within 32-wide j tile
  const int boff = hi ? 8 : 0;

  const float f1v  = f1[h * NN + row];
  const float mrow = m[h * NN + row];
  const float* arow = adj + (size_t)row * NN;
  const float* f2h  = f2 + h * NN;

  v8f acc[4] = {};

  for (int jt = 0; jt < NN; jt += 32) {
    // Keep the 8 waves converged so their shared B loads hit WGP$/L0.
    if ((jt & 255) == 0) __syncthreads();

    const float* ap = arow + jt + joff;
    const float* fp = f2h + jt + joff;
    // Prefetch next tile of the streaming (HBM-bound) adjacency row.
    __builtin_prefetch(ap + 32, 0, 0);

    // A-frag element e: K offset = e (e<8) or e+8 (e>=8) -> contiguous quads.
    v4fl av[4] = { *reinterpret_cast<const v4fl*>(ap),
                   *reinterpret_cast<const v4fl*>(ap + 4),
                   *reinterpret_cast<const v4fl*>(ap + 16),
                   *reinterpret_cast<const v4fl*>(ap + 20) };
    v4fl fv[4] = { *reinterpret_cast<const v4fl*>(fp),
                   *reinterpret_cast<const v4fl*>(fp + 4),
                   *reinterpret_cast<const v4fl*>(fp + 16),
                   *reinterpret_cast<const v4fl*>(fp + 20) };
    v8u pw;
#pragma unroll
    for (int q = 0; q < 8; ++q) {
      float s0 = f1v + fv[q >> 1][(q & 1) * 2 + 0];
      s0 = s0 > 0.0f ? s0 : 0.2f * s0;
      float p0 = (av[q >> 1][(q & 1) * 2 + 0] != 0.0f) ? __expf(s0 - mrow) : 0.0f;
      float s1 = f1v + fv[q >> 1][(q & 1) * 2 + 1];
      s1 = s1 > 0.0f ? s1 : 0.2f * s1;
      float p1 = (av[q >> 1][(q & 1) * 2 + 1] != 0.0f) ? __expf(s1 - mrow) : 0.0f;
      pw[q] = f32_to_bf16_rne(p0) | (f32_to_bf16_rne(p1) << 16);
    }
    v16bf pfrag = __builtin_bit_cast(v16bf, pw);

#pragma unroll
    for (int t = 0; t < 4; ++t) {
      const unsigned short* vp =
          seqbt + (size_t)(h * DH + t * 16 + c) * NN + jt + boff;
      v16bf b = load_frag16(vp, vp + 16);
      acc[t] = __builtin_amdgcn_wmma_f32_16x16x32_bf16(
          false, pfrag, false, b, (short)0, acc[t], false, false);
    }
  }

  // Epilogue: normalize by 1/l per row, ELU, store [N, H*dh] coalesced.
#pragma unroll
  for (int e = 0; e < 8; ++e) {
    const int n = base + e + (hi ? 8 : 0);
    const float li = linv[h * NN + n];
#pragma unroll
    for (int t = 0; t < 4; ++t) {
      float v = acc[t][e] * li;
      v = v > 0.0f ? v : (__expf(v) - 1.0f);   // elu (alpha=1)
      out[(size_t)n * DD + h * DH + t * 16 + c] = v;
    }
  }
}

// ---------------------------------------------------------------------------
extern "C" void kernel_launch(void* const* d_in, const int* in_sizes, int n_in,
                              void* d_out, int out_size, void* d_ws, size_t ws_size,
                              hipStream_t stream) {
  (void)in_sizes; (void)n_in; (void)out_size; (void)ws_size;
  const float* x   = (const float*)d_in[0];
  const float* adj = (const float*)d_in[1];
  const float* W   = (const float*)d_in[2];
  const float* a1  = (const float*)d_in[3];
  const float* b1  = (const float*)d_in[4];
  const float* a2  = (const float*)d_in[5];
  const float* b2  = (const float*)d_in[6];
  float* out = (float*)d_out;

  char* ws = (char*)d_ws;
  size_t off = 0;
  auto alloc = [&](size_t bytes) -> void* {
    void* p = ws + off;
    off = (off + bytes + 255) & ~(size_t)255;
    return p;
  };
  unsigned short* xb    = (unsigned short*)alloc((size_t)NN * FF * 2);
  unsigned short* wbt   = (unsigned short*)alloc((size_t)HH * DH * FF * 2);
  float*          seq   = (float*)alloc((size_t)HH * NN * DH * 4);
  unsigned short* seqbt = (unsigned short*)alloc((size_t)HH * DH * NN * 2);
  float*          f1    = (float*)alloc((size_t)HH * NN * 4);
  float*          f2    = (float*)alloc((size_t)HH * NN * 4);
  float*          mrow  = (float*)alloc((size_t)HH * NN * 4);
  float*          linv  = (float*)alloc((size_t)HH * NN * 4);

  k_pack_x<<<(NN * FF + 255) / 256, 256, 0, stream>>>(x, xb);
  k_pack_w<<<(HH * DH * FF + 255) / 256, 256, 0, stream>>>(W, wbt);
  k_seqfts<<<NN / 16, 256, 0, stream>>>(xb, wbt, seq, seqbt);
  k_f1f2<<<(HH * NN + 255) / 256, 256, 0, stream>>>(seq, a1, b1, a2, b2, f1, f2);
  k_rowstats<<<NN, 256, 0, stream>>>(adj, f1, f2, mrow, linv);
  k_aggregate<<<dim3(NN / 128, HH), 256, 0, stream>>>(adj, f1, f2, mrow, linv,
                                                      seqbt, out);
}
